// SCAD_26955214750132
// MI455X (gfx1250) — matrix operand order repaired and checked by reading, not compiled, not run on hardware
//
#include <hip/hip_runtime.h>
#include <hip/hip_bf16.h>

// ---------------------------------------------------------------------------
// SCAD forward for MI455X (gfx1250, wave32, WMMA).
// Heavy GEMMs (GRU over 12800 seqs x 50 steps; 100k-item scoring) run on
// v_wmma_f32_16x16x32_bf16. Everything else is VALU glue.
// Round 3: asm memory fences inside the GRU step stop MachineLICM from
// hoisting the 48 loop-invariant weight-fragment loads (which previously
// spilled to scratch); each j sub-body now streams its 12 fragment loads
// directly into its 12 WMMAs.
// ---------------------------------------------------------------------------

typedef __bf16 bf16_t;
typedef bf16_t v16bf __attribute__((ext_vector_type(16)));
typedef float  v8f   __attribute__((ext_vector_type(8)));

#define NB_     256
#define NS_     50
#define ND_     64
#define NITEM_  100000
#define NSEQ_   12800           // 256 main + 256*49 neighbors
#define NTILE_  6250            // 100000 / 16

// workspace byte offsets
#define OFF_HS   ((size_t)0)                    // 12800*50*64 f32 = 163,840,000
#define OFF_USR  ((size_t)163840000)            // 256*64 f32
#define OFF_NEI  ((size_t)163905536)            // 256*50*64 f32
#define OFF_KK   ((size_t)167182336)
#define OFF_VV   ((size_t)170459136)
#define OFF_A1   ((size_t)173735936)            // 256*64 f32
#define OFF_A2   ((size_t)173801472)            // 256*4*64 f32
#define OFF_WPK  ((size_t)174063616)            // 1536*16 bf16 (GRU weight frags)
#define OFF_EPK  ((size_t)174112768)            // 6250*2*32*16 bf16 (emb B frags)

__device__ __forceinline__ float sigmf(float x) { return 1.f / (1.f + expf(-x)); }

__device__ __forceinline__ v8f wmma_bf16(v16bf a, v16bf b, v8f c) {
  return __builtin_amdgcn_wmma_f32_16x16x32_bf16(false, a, false, b, (short)0, c,
                                                 false, false);
}

// Scheduling fence: conservative memory clobber. Blocks LICM hoisting of the
// loop-invariant fragment loads and keeps each sub-body's loads local.
__device__ __forceinline__ void sched_fence() { asm volatile("" ::: "memory"); }

// A-matrix 16x32 bf16 fragment: lane row = lane%16,
// half0 elems = K 0..15 (+32c), half1 = K 16..31 (+32c).
__device__ __forceinline__ v16bf packA(const float* p, int c, int half) {
  int f0 = 32 * c + 16 * half;
  int f1 = f0 + 8;
  v16bf r;
#pragma unroll
  for (int i = 0; i < 8; ++i) r[i] = (bf16_t)p[f0 + i];
#pragma unroll
  for (int i = 0; i < 8; ++i) r[8 + i] = (bf16_t)p[f1 + i];
  return r;
}

// B-matrix 32x16 bf16 fragment: lane col = lane%16,
// half0 = K 0..15 (+32c), half1 = K 16..31 (+32c) (same contiguous k order as packA).
__device__ __forceinline__ v16bf packB(const float* p, int c, int half) {
  int f0 = 32 * c + 16 * half;
  v16bf r;
#pragma unroll
  for (int i = 0; i < 16; ++i) r[i] = (bf16_t)p[f0 + i];
  return r;
}

// --------------------------- setup: pack GRU weights ------------------------
__global__ void k_pack_w(const float* __restrict__ Wih, const float* __restrict__ Whh,
                         bf16_t* __restrict__ wpk) {
  int e = blockIdx.x * blockDim.x + threadIdx.x;   // ((mat*12+mt)*2+c)*32+lane
  if (e >= 2 * 12 * 2 * 32) return;
  int lane = e & 31;
  int c    = (e >> 5) & 1;
  int mt   = (e >> 6) % 12;
  int mat  = (e >> 6) / 12;
  int row  = mt * 16 + (lane & 15);
  int half = lane >> 4;
  const float* src = (mat ? Whh : Wih) + row * 64;
  bf16_t* dst = wpk + (size_t)e * 16;
  int f0 = 32 * c + 16 * half, f1 = f0 + 8;
#pragma unroll
  for (int i = 0; i < 8; ++i) dst[i] = (bf16_t)src[f0 + i];
#pragma unroll
  for (int i = 0; i < 8; ++i) dst[8 + i] = (bf16_t)src[f1 + i];
}

// --------------------------- setup: pack emb as B frags ---------------------
__global__ void k_pack_emb(const float* __restrict__ emb, bf16_t* __restrict__ epk) {
  int e = blockIdx.x * blockDim.x + threadIdx.x;   // (nt*2+c)*32+lane
  if (e >= NTILE_ * 2 * 32) return;
  int lane = e & 31;
  int c    = (e >> 5) & 1;
  int nt   = e >> 6;
  int item = nt * 16 + (lane & 15);
  int half = lane >> 4;
  const float* src = emb + (size_t)item * 64 + 32 * c + 16 * half;
  bf16_t* dst = epk + (size_t)e * 16;
#pragma unroll
  for (int i = 0; i < 16; ++i) dst[i] = (bf16_t)src[i];
}

// --------------------------- GRU over all 12800 sequences -------------------
// wave handles 16 sequences; g^T = W_ih x_t^T + W_hh h^T computed tile-by-tile
// (rows = gate dim, cols = sequences). Per feature tile j only 4 accumulator
// fragments are live (r, z, gi_n, gh_n); gate math is applied immediately.
// h lives in D-layout fragments; B_h repack = lane-half swap (shfl_xor 16).
__global__ void __launch_bounds__(256, 1)
k_gru(const int* __restrict__ td, const int* __restrict__ tnd,
      const float* __restrict__ emb, const bf16_t* __restrict__ wpk,
      const float* __restrict__ b_ih, const float* __restrict__ b_hh,
      float* __restrict__ hs) {
  __shared__ float bsum[128], bin_[64], bhn_[64];
  int tid = threadIdx.x;
  if (tid < 128) bsum[tid] = b_ih[tid] + b_hh[tid];
  else if (tid < 192) { bin_[tid - 128] = b_ih[tid]; bhn_[tid - 128] = b_hh[tid]; }
  __syncthreads();

  int w = tid >> 5, lane = tid & 31;
  int col = lane & 15, half = lane >> 4;
  int seq = (blockIdx.x * 8 + w) * 16 + col;
  const int* ip = (seq < 256) ? (td + seq * 50) : (tnd + (size_t)(seq - 256) * 50);
  const v16bf* wf = reinterpret_cast<const v16bf*>(wpk);
  float* hout = hs + (size_t)seq * (NS_ * ND_);

  v8f h[4];
#pragma unroll
  for (int j = 0; j < 4; ++j) h[j] = v8f{};

  int mbh = 8 * half;
  for (int t = 0; t < NS_; ++t) {
    const float* er = emb + (size_t)ip[t] * 64;
    v16bf bx0 = packB(er, 0, half);
    v16bf bx1 = packB(er, 1, half);
    // repack h (D layout) into B layout: half-swap via shfl_xor(16)
    v16bf bh0, bh1;
#pragma unroll
    for (int e = 0; e < 8; ++e) {
      float s0 = __shfl_xor(h[0][e], 16, 32);
      float s1 = __shfl_xor(h[1][e], 16, 32);
      float s2 = __shfl_xor(h[2][e], 16, 32);
      float s3 = __shfl_xor(h[3][e], 16, 32);
      bh0[e]     = (bf16_t)(half ? s1 : h[0][e]);
      bh0[8 + e] = (bf16_t)(half ? h[1][e] : s0);
      bh1[e]     = (bf16_t)(half ? s3 : h[2][e]);
      bh1[8 + e] = (bf16_t)(half ? h[3][e] : s2);
    }
#pragma unroll
    for (int j = 0; j < 4; ++j) {
      sched_fence();   // keep this sub-body's 12 fragment loads in place
      // r gate: weight row tile j (W_ih and W_hh fused into one accumulator)
      v8f racc = {};
      racc = wmma_bf16(wf[((0 * 12 + j) * 2 + 0) * 32 + lane], bx0, racc);
      racc = wmma_bf16(wf[((0 * 12 + j) * 2 + 1) * 32 + lane], bx1, racc);
      racc = wmma_bf16(wf[((1 * 12 + j) * 2 + 0) * 32 + lane], bh0, racc);
      racc = wmma_bf16(wf[((1 * 12 + j) * 2 + 1) * 32 + lane], bh1, racc);
      // z gate: weight row tile 4+j
      v8f zacc = {};
      zacc = wmma_bf16(wf[((0 * 12 + 4 + j) * 2 + 0) * 32 + lane], bx0, zacc);
      zacc = wmma_bf16(wf[((0 * 12 + 4 + j) * 2 + 1) * 32 + lane], bx1, zacc);
      zacc = wmma_bf16(wf[((1 * 12 + 4 + j) * 2 + 0) * 32 + lane], bh0, zacc);
      zacc = wmma_bf16(wf[((1 * 12 + 4 + j) * 2 + 1) * 32 + lane], bh1, zacc);
      // n gate: weight row tile 8+j, gi and gh kept separate
      v8f gi = {};
      gi = wmma_bf16(wf[((0 * 12 + 8 + j) * 2 + 0) * 32 + lane], bx0, gi);
      gi = wmma_bf16(wf[((0 * 12 + 8 + j) * 2 + 1) * 32 + lane], bx1, gi);
      v8f gh = {};
      gh = wmma_bf16(wf[((1 * 12 + 8 + j) * 2 + 0) * 32 + lane], bh0, gh);
      gh = wmma_bf16(wf[((1 * 12 + 8 + j) * 2 + 1) * 32 + lane], bh1, gh);
      int mb = 16 * j + mbh;
#pragma unroll
      for (int e = 0; e < 8; ++e) {
        float r = sigmf(racc[e] + bsum[mb + e]);
        float z = sigmf(zacc[e] + bsum[64 + mb + e]);
        float n = tanhf(gi[e] + bin_[mb + e] + r * (gh[e] + bhn_[mb + e]));
        h[j][e] = (1.f - z) * n + z * h[j][e];
      }
      float* hp = hout + t * 64 + mb;
#pragma unroll
      for (int e = 0; e < 8; ++e) hp[e] = h[j][e];
    }
  }
}

// --------------------------- usr attention (rows 0..255) --------------------
__global__ void k_usratt(const float* __restrict__ hs, const float* __restrict__ w1,
                         const float* __restrict__ b1, const float* __restrict__ w2,
                         const float* __restrict__ b2p, float* __restrict__ usr,
                         float* __restrict__ nei) {
  int wv = (blockIdx.x * blockDim.x + threadIdx.x) >> 5;
  int lane = threadIdx.x & 31;
  if (wv >= NB_) return;
  const float* base = hs + (size_t)wv * (NS_ * ND_);
  const float* wr = w1 + lane * 64;
  float bb = b1[lane];
  const float* hl = base + 49 * 64;
  float an = bb;
  for (int f = 0; f < 64; ++f) an += wr[f] * hl[f];
  float w2v = w2[lane], b2v = b2p[0];
  float acc0 = 0.f, acc1 = 0.f;
  for (int t = 0; t < NS_; ++t) {
    const float* ht = base + t * 64;
    float ai = bb;
    for (int f = 0; f < 64; ++f) ai += wr[f] * ht[f];
    float s = sigmf(an + ai) * w2v;
    for (int o = 16; o; o >>= 1) s += __shfl_xor(s, o, 32);
    float att = s + b2v;
    acc0 += att * ht[lane];
    acc1 += att * ht[lane + 32];
  }
  usr[wv * 64 + lane] = acc0;
  usr[wv * 64 + lane + 32] = acc1;
  nei[(size_t)wv * 3200 + lane] = acc0;          // nei row 0 = usr
  nei[(size_t)wv * 3200 + lane + 32] = acc1;
}

// --------------------------- neighbor attention (T=51, last entry = usr) ----
__global__ void k_neiatt(const float* __restrict__ hs, const float* __restrict__ usr,
                         const float* __restrict__ w1, const float* __restrict__ b1,
                         const float* __restrict__ w2, const float* __restrict__ b2p,
                         float* __restrict__ nei) {
  int wv = (blockIdx.x * blockDim.x + threadIdx.x) >> 5;
  int lane = threadIdx.x & 31;
  if (wv >= NB_ * (NS_ - 1)) return;
  int b = wv / 49, jj = wv % 49;
  const float* base = hs + (size_t)(256 + wv) * (NS_ * ND_);
  const float* up = usr + b * 64;
  const float* wr = w1 + lane * 64;
  float bb = b1[lane];
  float an = bb;
  for (int f = 0; f < 64; ++f) an += wr[f] * up[f];
  float w2v = w2[lane], b2v = b2p[0];
  float acc0 = 0.f, acc1 = 0.f;
  for (int t = 0; t < 51; ++t) {
    const float* ht = (t < 50) ? base + t * 64 : up;
    float ai = bb;
    for (int f = 0; f < 64; ++f) ai += wr[f] * ht[f];
    float s = sigmf(an + ai) * w2v;
    for (int o = 16; o; o >>= 1) s += __shfl_xor(s, o, 32);
    float att = s + b2v;
    acc0 += att * ht[lane];
    acc1 += att * ht[lane + 32];
  }
  nei[(size_t)b * 3200 + (jj + 1) * 64 + lane] = acc0;
  nei[(size_t)b * 3200 + (jj + 1) * 64 + lane + 32] = acc1;
}

// --------------------------- self-attn + LN1 + kk/vv ------------------------
__global__ void k_selfatt(float* __restrict__ nei,
                          const float* __restrict__ Wq, const float* __restrict__ Wk,
                          const float* __restrict__ Wv, const float* __restrict__ l1g,
                          const float* __restrict__ l1b, const float* __restrict__ k1W,
                          const float* __restrict__ v1W, float* __restrict__ kk,
                          float* __restrict__ vv) {
  __shared__ float xb[3200], qs[3200], ks[3200], vs[3200];
  int b = blockIdx.x, tid = threadIdx.x;
  for (int i = tid; i < 3200; i += 256) xb[i] = nei[(size_t)b * 3200 + i];
  __syncthreads();
  for (int i = tid; i < 3200; i += 256) {
    int row = i >> 6, d = i & 63;
    const float* x = xb + row * 64;
    const float* wq = Wq + d * 64;
    const float* wk = Wk + d * 64;
    const float* wv = Wv + d * 64;
    float aq = 0.f, ak = 0.f, av = 0.f;
    for (int f = 0; f < 64; ++f) { float xv = x[f]; aq += xv * wq[f]; ak += xv * wk[f]; av += xv * wv[f]; }
    qs[i] = aq; ks[i] = ak; vs[i] = av;
  }
  __syncthreads();
  if (tid < NS_) {
    int s = tid;
    float sc[50];
    float mx = -1e30f;
    for (int t = 0; t < 50; ++t) {
      const float* qp = qs + s * 64;
      const float* kp = ks + t * 64;
      float a = 0.f;
      for (int f = 0; f < 64; ++f) a += qp[f] * kp[f];
      a *= 0.125f;
      sc[t] = a;
      mx = fmaxf(mx, a);
    }
    float se = 0.f;
    for (int t = 0; t < 50; ++t) { sc[t] = expf(sc[t] - mx); se += sc[t]; }
    float inv = 1.f / se;
    float ov[64];
    float m = 0.f;
    for (int d = 0; d < 64; ++d) {
      float a = 0.f;
      for (int t = 0; t < 50; ++t) a += sc[t] * vs[t * 64 + d];
      float y = xb[s * 64 + d] + a * inv;
      ov[d] = y; m += y;
    }
    m *= (1.f / 64.f);
    float var = 0.f;
    for (int d = 0; d < 64; ++d) { float dd = ov[d] - m; var += dd * dd; }
    var *= (1.f / 64.f);
    float is = rsqrtf(var + 1e-5f);
    for (int d = 0; d < 64; ++d) ov[d] = (ov[d] - m) * is * l1g[d] + l1b[d];
    float* np = nei + (size_t)b * 3200 + s * 64;
    for (int d = 0; d < 64; ++d) np[d] = ov[d];
    float* kp2 = kk + (size_t)b * 3200 + s * 64;
    float* vp2 = vv + (size_t)b * 3200 + s * 64;
    for (int d = 0; d < 64; ++d) {
      const float* w1 = k1W + d * 64;
      const float* w2 = v1W + d * 64;
      float a = 0.f, c2 = 0.f;
      for (int f = 0; f < 64; ++f) { a += ov[f] * w1[f]; c2 += ov[f] * w2[f]; }
      kp2[d] = a; vp2[d] = c2;
    }
  }
}

// --------------------------- single dynamic routing -> a1 -------------------
__global__ void k_route1(const float* __restrict__ nei, const float* __restrict__ kk,
                         const float* __restrict__ vv, const float* __restrict__ q1W,
                         const float* __restrict__ l2g, const float* __restrict__ l2b,
                         float* __restrict__ a1) {
  __shared__ float nb[3200], kb[3200], vb[3200];
  __shared__ float bvec[50], qv[50], aw[64], oo[50], cc[50], red[64];
  __shared__ float s_max, s_sum, s_fac;
  int b = blockIdx.x, tid = threadIdx.x;
  for (int i = tid; i < 3200; i += 64) {
    nb[i] = nei[(size_t)b * 3200 + i];
    kb[i] = kk[(size_t)b * 3200 + i];
    vb[i] = vv[(size_t)b * 3200 + i];
  }
  if (tid < 50) bvec[tid] = 0.f;
  __syncthreads();
  float a1d = 0.f;
  for (int it = 0; it < 3; ++it) {
    if (tid < 50) {
      float a = 0.f;
      for (int s = 0; s < 50; ++s) a += bvec[s] * q1W[tid * 50 + s];
      qv[tid] = a;
    }
    __syncthreads();
    float e = 0.f;
    for (int s = 0; s < 50; ++s) e += qv[s] * kb[s * 64 + tid];
    e *= 0.125f;
    red[tid] = e; __syncthreads();
    if (tid == 0) { float m = red[0]; for (int i = 1; i < 64; ++i) m = fmaxf(m, red[i]); s_max = m; }
    __syncthreads();
    float ex = expf(e - s_max);
    red[tid] = ex; __syncthreads();
    if (tid == 0) { float s = 0.f; for (int i = 0; i < 64; ++i) s += red[i]; s_sum = s; }
    __syncthreads();
    aw[tid] = ex / s_sum;
    __syncthreads();
    if (tid < 50) {
      float a = 0.f;
      for (int d = 0; d < 64; ++d) a += aw[d] * vb[tid * 64 + d];
      oo[tid] = a;
    }
    __syncthreads();
    if (tid == 0) {
      float tv[50], sb[50], pv[50];
      float m = 0.f;
      for (int s = 0; s < 50; ++s) { tv[s] = bvec[s] + oo[s]; m += tv[s]; }
      m *= 0.02f;
      float var = 0.f;
      for (int s = 0; s < 50; ++s) { float d = tv[s] - m; var += d * d; }
      var *= 0.02f;
      float is = rsqrtf(var + 1e-5f);
      float bm = bvec[0];
      for (int s = 1; s < 50; ++s) bm = fmaxf(bm, bvec[s]);
      float bs = 0.f;
      for (int s = 0; s < 50; ++s) { sb[s] = expf(bvec[s] - bm); bs += sb[s]; }
      float pm = -1e30f;
      for (int s = 0; s < 50; ++s) {
        float ln = (tv[s] - m) * is * l2g[s] + l2b[s];
        pv[s] = (sb[s] / bs) * ln;
        pm = fmaxf(pm, pv[s]);
      }
      float ps = 0.f;
      for (int s = 0; s < 50; ++s) { pv[s] = expf(pv[s] - pm); ps += pv[s]; }
      for (int s = 0; s < 50; ++s) cc[s] = pv[s] / ps;
    }
    __syncthreads();
    float ad = 0.f;
    for (int s = 0; s < 50; ++s) ad += cc[s] * nb[s * 64 + tid];
    red[tid] = ad; __syncthreads();
    if (tid == 0) {
      float sn = 0.f;
      for (int i = 0; i < 64; ++i) sn += red[i] * red[i];
      s_fac = (sn / (1.f + sn)) / (sqrtf(sn) + 1e-8f);
    }
    __syncthreads();
    a1d = ad * s_fac;
    red[tid] = a1d; __syncthreads();
    if (tid < 50) {
      float u = 0.f;
      for (int d = 0; d < 64; ++d) u += red[d] * nb[tid * 64 + d];
      bvec[tid] += u;
    }
    __syncthreads();
  }
  a1[b * 64 + tid] = a1d;
}

// --------------------------- multi routing (K=4) -> a2 ----------------------
__global__ void k_route2(const float* __restrict__ nei, const float* __restrict__ kk,
                         const float* __restrict__ vv, const float* __restrict__ q1W,
                         const float* __restrict__ l2g, const float* __restrict__ l2b,
                         const float* __restrict__ binit, float* __restrict__ a2) {
  __shared__ float nb[3200], kb[3200], vb[3200];
  __shared__ float b2[4][50], qv[50], aw[64], oo[50], cc[50], red[64];
  __shared__ float s_max, s_sum, s_fac;
  int b = blockIdx.x, tid = threadIdx.x;
  for (int i = tid; i < 3200; i += 64) {
    nb[i] = nei[(size_t)b * 3200 + i];
    kb[i] = kk[(size_t)b * 3200 + i];
    vb[i] = vv[(size_t)b * 3200 + i];
  }
  for (int i = tid; i < 200; i += 64) b2[i / 50][i % 50] = binit[i];
  __syncthreads();
  float a2d[4] = {0.f, 0.f, 0.f, 0.f};
  for (int it = 0; it < 3; ++it) {
    for (int k = 0; k < 4; ++k) {
      if (tid < 50) {
        float a = 0.f;
        for (int s = 0; s < 50; ++s) a += b2[k][s] * q1W[tid * 50 + s];
        qv[tid] = a;
      }
      __syncthreads();
      float e = 0.f;
      for (int s = 0; s < 50; ++s) e += qv[s] * kb[s * 64 + tid];
      e *= 0.125f;
      red[tid] = e; __syncthreads();
      if (tid == 0) { float m = red[0]; for (int i = 1; i < 64; ++i) m = fmaxf(m, red[i]); s_max = m; }
      __syncthreads();
      float ex = expf(e - s_max);
      red[tid] = ex; __syncthreads();
      if (tid == 0) { float s = 0.f; for (int i = 0; i < 64; ++i) s += red[i]; s_sum = s; }
      __syncthreads();
      aw[tid] = ex / s_sum;
      __syncthreads();
      if (tid < 50) {
        float a = 0.f;
        for (int d = 0; d < 64; ++d) a += aw[d] * vb[tid * 64 + d];
        oo[tid] = a;
      }
      __syncthreads();
      if (tid == 0) {
        float tv[50], sb[50], pv[50];
        float m = 0.f;
        for (int s = 0; s < 50; ++s) { tv[s] = b2[k][s] + oo[s]; m += tv[s]; }
        m *= 0.02f;
        float var = 0.f;
        for (int s = 0; s < 50; ++s) { float d = tv[s] - m; var += d * d; }
        var *= 0.02f;
        float is = rsqrtf(var + 1e-5f);
        float bm = b2[k][0];
        for (int s = 1; s < 50; ++s) bm = fmaxf(bm, b2[k][s]);
        float bs = 0.f;
        for (int s = 0; s < 50; ++s) { sb[s] = expf(b2[k][s] - bm); bs += sb[s]; }
        float pm = -1e30f;
        for (int s = 0; s < 50; ++s) {
          float ln = (tv[s] - m) * is * l2g[s] + l2b[s];
          pv[s] = (sb[s] / bs) * ln;
          pm = fmaxf(pm, pv[s]);
        }
        float ps = 0.f;
        for (int s = 0; s < 50; ++s) { pv[s] = expf(pv[s] - pm); ps += pv[s]; }
        for (int s = 0; s < 50; ++s) cc[s] = pv[s] / ps;
      }
      __syncthreads();
      float ad = 0.f;
      for (int s = 0; s < 50; ++s) ad += cc[s] * nb[s * 64 + tid];
      red[tid] = ad; __syncthreads();
      if (tid == 0) {
        float sn = 0.f;
        for (int i = 0; i < 64; ++i) sn += red[i] * red[i];
        s_fac = (sn / (1.f + sn)) / (sqrtf(sn) + 1e-8f);
      }
      __syncthreads();
      a2d[k] = ad * s_fac;
      red[tid] = a2d[k]; __syncthreads();
      if (it < 2 && tid < 50) {
        float u = 0.f;
        for (int d = 0; d < 64; ++d) u += red[d] * nb[tid * 64 + d];
        b2[k][tid] += u;
      }
      __syncthreads();
    }
  }
#pragma unroll
  for (int k = 0; k < 4; ++k) a2[b * 256 + k * 64 + tid] = a2d[k];
}

// --------------------------- scoring: yita*a1@E^T + (1-yita)*max_k a2@E^T ---
#define SC_NBPB 98   // blocks per batch-tile; 98*8 = 784 waves per batch-tile
__global__ void __launch_bounds__(256, 1)
k_score(const float* __restrict__ a1, const float* __restrict__ a2,
        const bf16_t* __restrict__ epk, const float* __restrict__ yita,
        float* __restrict__ out) {
  int w = threadIdx.x >> 5, lane = threadIdx.x & 31;
  int bt  = blockIdx.x / SC_NBPB;
  int blk = blockIdx.x % SC_NBPB;
  int wv  = blk * 8 + w;
  int half = lane >> 4, lidx = lane & 15;
  float yv = *yita;
  int brow = bt * 16 + lidx;
  v16bf a1f0 = packA(a1 + brow * 64, 0, half);
  v16bf a1f1 = packA(a1 + brow * 64, 1, half);
  v16bf a2f[4][2];
#pragma unroll
  for (int k = 0; k < 4; ++k) {
    const float* p = a2 + brow * 256 + k * 64;
    a2f[k][0] = packA(p, 0, half);
    a2f[k][1] = packA(p, 1, half);
  }
  const v16bf* ef = reinterpret_cast<const v16bf*>(epk);
  for (int nt = wv; nt < NTILE_; nt += SC_NBPB * 8) {
    if (nt + SC_NBPB * 8 < NTILE_)
      __builtin_prefetch(&ef[(size_t)(nt + SC_NBPB * 8) * 2 * 32 + lane], 0, 1);
    v16bf b0 = ef[(size_t)(nt * 2 + 0) * 32 + lane];
    v16bf b1 = ef[(size_t)(nt * 2 + 1) * 32 + lane];
    v8f s1 = {};
    s1 = wmma_bf16(a1f0, b0, s1);
    s1 = wmma_bf16(a1f1, b1, s1);
    v8f mx;
    {
      v8f s = {};
      s = wmma_bf16(a2f[0][0], b0, s);
      s = wmma_bf16(a2f[0][1], b1, s);
      mx = s;
    }
#pragma unroll
    for (int k = 1; k < 4; ++k) {
      v8f s = {};
      s = wmma_bf16(a2f[k][0], b0, s);
      s = wmma_bf16(a2f[k][1], b1, s);
#pragma unroll
      for (int r = 0; r < 8; ++r) mx[r] = fmaxf(mx[r], s[r]);
    }
    int n0 = nt * 16 + lidx;
    int rowb = bt * 16 + 8 * half;
#pragma unroll
    for (int r = 0; r < 8; ++r) {
      float v = yv * s1[r] + (1.f - yv) * mx[r];
      out[(size_t)(rowb + r) * NITEM_ + n0] = v;
    }
  }
}

// ---------------------------------------------------------------------------
extern "C" void kernel_launch(void* const* d_in, const int* in_sizes, int n_in,
                              void* d_out, int out_size, void* d_ws, size_t ws_size,
                              hipStream_t stream) {
  (void)in_sizes; (void)n_in; (void)out_size; (void)ws_size;
  const int*   td   = (const int*)d_in[0];
  const int*   tnd  = (const int*)d_in[1];
  const float* emb  = (const float*)d_in[2];
  const float* Wih  = (const float*)d_in[3];
  const float* Whh  = (const float*)d_in[4];
  const float* bih  = (const float*)d_in[5];
  const float* bhh  = (const float*)d_in[6];
  const float* a1W  = (const float*)d_in[7];
  const float* a1b  = (const float*)d_in[8];
  const float* a2W  = (const float*)d_in[9];
  const float* a2b  = (const float*)d_in[10];
  const float* Wq   = (const float*)d_in[11];
  const float* Wk   = (const float*)d_in[12];
  const float* Wv   = (const float*)d_in[13];
  const float* l1g  = (const float*)d_in[14];
  const float* l1b  = (const float*)d_in[15];
  const float* l2g  = (const float*)d_in[16];
  const float* l2b  = (const float*)d_in[17];
  const float* q1W  = (const float*)d_in[18];
  const float* k1W  = (const float*)d_in[19];
  const float* v1W  = (const float*)d_in[20];
  const float* yita = (const float*)d_in[21];
  const float* bini = (const float*)d_in[22];

  char* ws = (char*)d_ws;
  float*  hs  = (float*)(ws + OFF_HS);
  float*  usr = (float*)(ws + OFF_USR);
  float*  nei = (float*)(ws + OFF_NEI);
  float*  kk  = (float*)(ws + OFF_KK);
  float*  vv  = (float*)(ws + OFF_VV);
  float*  a1  = (float*)(ws + OFF_A1);
  float*  a2  = (float*)(ws + OFF_A2);
  bf16_t* wpk = (bf16_t*)(ws + OFF_WPK);
  bf16_t* epk = (bf16_t*)(ws + OFF_EPK);
  float*  out = (float*)d_out;

  k_pack_w  <<<6,    256, 0, stream>>>(Wih, Whh, wpk);
  k_pack_emb<<<1563, 256, 0, stream>>>(emb, epk);
  k_gru     <<<100,  256, 0, stream>>>(td, tnd, emb, wpk, bih, bhh, hs);
  k_usratt  <<<32,   256, 0, stream>>>(hs, a1W, a1b, a2W, a2b, usr, nei);
  k_neiatt  <<<1568, 256, 0, stream>>>(hs, usr, a1W, a1b, a2W, a2b, nei);
  k_selfatt <<<256,  256, 0, stream>>>(nei, Wq, Wk, Wv, l1g, l1b, k1W, v1W, kk, vv);
  k_route1  <<<256,  64,  0, stream>>>(nei, kk, vv, q1W, l2g, l2b, a1);
  k_route2  <<<256,  64,  0, stream>>>(nei, kk, vv, q1W, l2g, l2b, bini, a2);
  k_score   <<<16 * SC_NBPB, 256, 0, stream>>>(a1, a2, epk, yita, out);
}
